// CausalMultiHeadSelfAttention_90649579750076
// MI455X (gfx1250) — matrix operand order repaired and checked by reading, not compiled, not run on hardware
//
#include <hip/hip_runtime.h>
#include <hip/hip_bf16.h>

// ---------------------------------------------------------------------------
// Causal MHSA on MI455X (gfx1250): bf16 WMMA (16x16x32, fp32 accum) for all
// three matmul stages, fused RoPE, flash-attention with online softmax.
// K tile staged by the Tensor Data Mover (TENSOR_LOAD_TO_LDS, TENSORcnt);
// V tile staged with GLOBAL_LOAD_ASYNC_TO_LDS_B128 (ASYNCcnt).
// ---------------------------------------------------------------------------

typedef __attribute__((ext_vector_type(16))) __bf16 v16bf;
typedef __attribute__((ext_vector_type(8)))  float  v8f;
typedef __attribute__((ext_vector_type(4)))  unsigned int u32x4;
typedef __attribute__((ext_vector_type(8)))  int   i32x8;
typedef __attribute__((ext_vector_type(4)))  int   i32x4;

#define D_MODEL 1024
#define SEQ     2048
#define BATCH   4
#define NHEADS  16
#define DHEAD   64
#define MTOT    (BATCH * SEQ)          // 8192
#define QKV_ELEMS ((size_t)BATCH * NHEADS * SEQ * DHEAD)   // 8388608

union AFrag { v16bf v; unsigned u[8]; };
union BFrag { v16bf v; unsigned u[8]; };

__device__ inline v8f vzero() {
    v8f z = {0.f, 0.f, 0.f, 0.f, 0.f, 0.f, 0.f, 0.f};
    return z;
}

__device__ inline v8f wmma_bf16(v16bf a, v16bf b, v8f c) {
    // D = A(16x32 bf16) * B(32x16 bf16) + C(16x16 f32)
    return __builtin_amdgcn_wmma_f32_16x16x32_bf16(
        /*neg_a=*/false, a, /*neg_b=*/false, b,
        /*c_mod=*/(short)0, c, /*reuse_a=*/false, /*reuse_b=*/false);
}

// A-fragment (16x32, 16-bit) K offset for VGPR j, lane-half h (ISA 7.12.2):
//   lanes 0-15:  j<4 -> K=2j     ; j>=4 -> K=2j+8
//   lanes 16-31: j<4 -> K=2j+8   ; j>=4 -> K=2j+16
__device__ inline int kOffA(int j, int half) {
    return 2 * j + 8 * (half + (j >= 4 ? 1 : 0));
}

__device__ inline unsigned pack2bf(float a, float b) {
    union { __bf16 h[2]; unsigned u; } p;
    p.h[0] = (__bf16)a;
    p.h[1] = (__bf16)b;
    return p.u;
}

// --- CDNA5 async global->LDS copy (tracked with ASYNCcnt) ------------------
// VFLAT encoding: VDST = LDS byte address, VADDR = 64-bit global address.
// Generic pointers to __shared__ carry the LDS offset in addr[31:0]
// (flat-aperture rule).
__device__ inline void async_cp_b128(unsigned lds_off, unsigned long long gaddr) {
    asm volatile("global_load_async_to_lds_b128 %0, %1, off"
                 :: "v"(lds_off), "v"(gaddr)
                 : "memory");
}
__device__ inline void wait_asynccnt0() {
    asm volatile("s_wait_asynccnt 0" ::: "memory");
}

// --- CDNA5 Tensor Data Mover: DMA a contiguous block into LDS --------------
// D# (ISA 8.3/8.4): 1-row 2D tile, data_size=8B, tile_dim0 = n8 (8-byte
// units).  All descriptor inputs here are wave-uniform -> SGPR groups.
__device__ inline void tdm_load_contig(unsigned lds_addr,
                                       unsigned long long gaddr,
                                       unsigned n8 /* 8-byte units */) {
    u32x4 g0;
    g0[0] = 1u;                                     // count=1 valid user D#
    g0[1] = lds_addr;                               // lds_addr [63:32]
    g0[2] = (unsigned)(gaddr & 0xffffffffu);        // global_addr lo
    g0[3] = (unsigned)(gaddr >> 32) | (2u << 30);   // addr hi | type=2(image)

    i32x8 g1;
    g1[0] = (int)(3u << 16);                        // wg_mask=0, data_size=8B
    g1[1] = (int)((n8 & 0xffffu) << 16);            // tensor_dim0[15:0]
    g1[2] = (int)(((n8 >> 16) & 0xffffu) | (1u << 16)); // td0 hi | tensor_dim1=1
    g1[3] = (int)(n8 << 16);                        // tile_dim0 = n8
    g1[4] = 1;                                      // tile_dim1=1, tile_dim2=0
    g1[5] = (int)n8;                                // tensor_dim0_stride lo
    g1[6] = 0;                                      // stride0 hi / stride1 lo
    g1[7] = 0;                                      // stride1 hi

    i32x4 z4 = {0, 0, 0, 0};
#if defined(__clang_major__) && (__clang_major__ >= 23)
    i32x8 z8 = {0, 0, 0, 0, 0, 0, 0, 0};
    __builtin_amdgcn_tensor_load_to_lds(g0, g1, z4, z4, z8, 0);
#else
    __builtin_amdgcn_tensor_load_to_lds(g0, g1, z4, z4, 0);
#endif
}
__device__ inline void wait_tensorcnt0() {
    __builtin_amdgcn_s_wait_tensorcnt(0);
}

// ---------------------------------------------------------------------------
// Tiled GEMM: C[M,N] = A[M,K] * Bw[N,K]^T  (both row-major, K contiguous).
// Block: 256 threads = 8 waves (2 x 4 wave grid), wave tile 32x32 (2x2 WMMA).
// Block tile 64(M) x 128(N), K staged in LDS in 32-wide bf16 chunks.
// MODE 0: store fp32 C.  MODE 1: fused RoPE + split into q/k/v bf16 ws
//         (V written in per-64-key transposed tiles: [b][h][kt][dh][key]).
// ---------------------------------------------------------------------------
template <int MODE>
__global__ __launch_bounds__(256)
void gemm_bf16_wmma(const float* __restrict__ A, const float* __restrict__ Bw,
                    float* __restrict__ Cout,
                    __bf16* __restrict__ qws, __bf16* __restrict__ kws,
                    __bf16* __restrict__ vws,
                    int M, int N, int K) {
    __shared__ __bf16 As[64][32];    // 4 KB
    __shared__ __bf16 Bs[128][32];   // 8 KB

    const int tid  = threadIdx.x;
    const int lane = tid & 31;
    const int w    = tid >> 5;
    const int wm   = w >> 2;         // 0..1
    const int wn   = w & 3;          // 0..3
    const int r16  = lane & 15;
    const int half = lane >> 4;
    const int blockM = blockIdx.y * 64;
    const int blockN = blockIdx.x * 128;

    v8f acc[2][2];
#pragma unroll
    for (int i = 0; i < 2; ++i)
#pragma unroll
        for (int j = 0; j < 2; ++j) acc[i][j] = vzero();

    const int nkb = K >> 5;
    for (int kb = 0; kb < nkb; ++kb) {
        __syncthreads();
        // ---- stage A tile (64 x 32 fp32 -> bf16) ----
#pragma unroll
        for (int it = 0; it < 2; ++it) {
            const int slot = tid + it * 256;         // 0..511
            const int r = slot >> 3, seg = slot & 7;
            const float4 v4 =
                *(const float4*)&A[(size_t)(blockM + r) * K + kb * 32 + seg * 4];
            *(unsigned*)&As[r][seg * 4]     = pack2bf(v4.x, v4.y);
            *(unsigned*)&As[r][seg * 4 + 2] = pack2bf(v4.z, v4.w);
        }
        // ---- stage B tile (128 x 32 fp32 -> bf16) ----
#pragma unroll
        for (int it = 0; it < 4; ++it) {
            const int slot = tid + it * 256;         // 0..1023
            const int r = slot >> 3, seg = slot & 7;
            const float4 v4 =
                *(const float4*)&Bw[(size_t)(blockN + r) * K + kb * 32 + seg * 4];
            *(unsigned*)&Bs[r][seg * 4]     = pack2bf(v4.x, v4.y);
            *(unsigned*)&Bs[r][seg * 4 + 2] = pack2bf(v4.z, v4.w);
        }
        // hint-prefetch next K chunk into cache (global_prefetch_b8)
        const int kb2 = (kb + 1 < nkb) ? kb + 1 : kb;
        __builtin_prefetch(&A[(size_t)(blockM + (tid >> 3)) * K + kb2 * 32], 0, 1);
        __builtin_prefetch(&Bw[(size_t)(blockN + (tid >> 3)) * K + kb2 * 32], 0, 1);
        __syncthreads();

        // ---- load fragments from LDS per ISA 7.12.2 layouts ----
        AFrag af[2];
#pragma unroll
        for (int i = 0; i < 2; ++i)
#pragma unroll
            for (int j = 0; j < 8; ++j)
                af[i].u[j] =
                    *(const unsigned*)&As[wm * 32 + 16 * i + r16][kOffA(j, half)];
        BFrag bfr[2];
#pragma unroll
        for (int jj = 0; jj < 2; ++jj)
#pragma unroll
            for (int j = 0; j < 8; ++j)
                bfr[jj].u[j] =
                    *(const unsigned*)&Bs[wn * 32 + 16 * jj + r16][kOffA(j, half)];

#pragma unroll
        for (int i = 0; i < 2; ++i)
#pragma unroll
            for (int jj = 0; jj < 2; ++jj)
                acc[i][jj] = wmma_bf16(af[i].v, bfr[jj].v, acc[i][jj]);
    }

    // ---- epilogue ----
#pragma unroll
    for (int i = 0; i < 2; ++i)
#pragma unroll
        for (int jj = 0; jj < 2; ++jj)
#pragma unroll
            for (int v = 0; v < 8; ++v) {
                const int mrow = blockM + wm * 32 + 16 * i + v + 8 * half;
                const int ncol = blockN + wn * 32 + 16 * jj + r16;
                float val = acc[i][jj][v];
                if (MODE == 0) {
                    Cout[(size_t)mrow * N + ncol] = val;
                } else {
                    // ncol in [0,3072): 0..1023 -> Q, 1024..2047 -> K, else V
                    const int section = ncol >> 10;
                    const int e  = ncol & 1023;
                    const int hh = e >> 6;
                    const int dh = e & 63;
                    const int bb  = mrow >> 11;     // / SEQ
                    const int pos = mrow & 2047;    // % SEQ
                    // RoPE pair partner lives in adjacent lane (adjacent N)
                    const float partner = __shfl_xor(val, 1, 32);
                    if (section < 2) {
                        // inv_freq = theta^(-(dh&~1)/64); ln(10000)=9.2103403...
                        const float inv =
                            __expf(-(float)(dh & ~1) * (9.210340371976184f / 64.0f));
                        const float ang = (float)pos * inv;
                        float sn, cs;
                        __sincosf(ang, &sn, &cs);
                        val = (dh & 1) ? (partner * sn + val * cs)
                                       : (val * cs - partner * sn);
                    }
                    const size_t head = (size_t)bb * NHEADS + hh;
                    if (section == 0) {
                        qws[(head * SEQ + pos) * DHEAD + dh] =
                            (__bf16)(val * 0.125f);          // fold 1/sqrt(64)
                    } else if (section == 1) {
                        kws[(head * SEQ + pos) * DHEAD + dh] = (__bf16)val;
                    } else {
                        // V stored transposed per 64-key tile: [kt][dh][key]
                        const int kt  = pos >> 6;
                        const int key = pos & 63;
                        vws[((head * (SEQ / 64) + kt) * DHEAD + dh) * 64 + key] =
                            (__bf16)val;
                    }
                }
            }
}

// ---------------------------------------------------------------------------
// Flash attention: one block per (64 query rows, head, batch).
// 128 threads = 4 waves; wave w owns query rows [qBase+16w, qBase+16w+16).
// Key loop only runs to the causal diagonal. K tile [key][dh] is DMA'd into
// LDS by the TDM (one tensor_load_to_lds issued by wave 0); V^T tile [dh][key]
// is staged with per-lane async b128 copies. S = Q K^T and O += P V on WMMA.
// ---------------------------------------------------------------------------
__global__ __launch_bounds__(128)
void flash_attn_wmma(const __bf16* __restrict__ qws,
                     const __bf16* __restrict__ kws,
                     const __bf16* __restrict__ vws,
                     float* __restrict__ yws) {
    __shared__ __bf16 Ks[64][64];       // 8 KB  [key][dh]
    __shared__ __bf16 VT[64][64];       // 8 KB  [dh][key]
    __shared__ __bf16 Ps[4][16][64];    // 8 KB  per-wave P staging

    const int tid  = threadIdx.x;
    const int lane = tid & 31;
    const int w    = tid >> 5;
    const int r16  = lane & 15;
    const int half = lane >> 4;
    const int qBase = blockIdx.x * 64;
    const int h = blockIdx.y;
    const int b = blockIdx.z;

    const size_t headOff = ((size_t)(b * NHEADS + h)) * SEQ * DHEAD;
    const __bf16* Qh = qws + headOff;
    const __bf16* Kh = kws + headOff;
    const __bf16* Vh = vws + headOff;   // transposed tiles: [kt][dh][key]

    const unsigned ksLds = (unsigned)(size_t)&Ks[0][0];
    const unsigned vtLds = (unsigned)(size_t)&VT[0][0];

    // Q fragments held in registers for the whole kernel (A layout, 16x64)
    AFrag qf[2];
    {
        const int qrow = qBase + w * 16 + r16;
#pragma unroll
        for (int f = 0; f < 2; ++f)
#pragma unroll
            for (int j = 0; j < 8; ++j) {
                const int dh = 32 * f + kOffA(j, half);
                qf[f].u[j] = *(const unsigned*)&Qh[(size_t)qrow * DHEAD + dh];
            }
    }

    v8f o[4];
#pragma unroll
    for (int t = 0; t < 4; ++t) o[t] = vzero();
    float mrow[8], lrow[8];
#pragma unroll
    for (int v = 0; v < 8; ++v) { mrow[v] = -1e30f; lrow[v] = 0.f; }

    const int nj = (qBase >> 6) + 1;     // causal: key tiles up to diagonal
    for (int j = 0; j < nj; ++j) {
        const int kB = j * 64;
        __syncthreads();   // previous iteration's LDS reads complete
        // ---- stage K (TDM) and V^T (async b128) tiles: 8 KB each ----
        {
            const unsigned long long kg =
                (unsigned long long)(const void*)(Kh + (size_t)kB * DHEAD);
            const unsigned long long vg =
                (unsigned long long)(const void*)(Vh + (size_t)j * DHEAD * 64);
            if (w == 0) {
                // one TDM op DMAs the whole contiguous 8 KB K tile
                tdm_load_contig(ksLds, kg, (64 * DHEAD * 2) / 8);
            }
#pragma unroll
            for (int it = 0; it < 4; ++it) {
                const unsigned byteOff = (unsigned)(tid + it * 128) * 16u;
                async_cp_b128(vtLds + byteOff, vg + byteOff);
            }
            wait_asynccnt0();
            if (w == 0) wait_tensorcnt0();
        }
        __syncthreads();

        // ---- S = Q K^T  (4 tiles of 16 keys; K-dim = dhead = 2x32) ----
        v8f s[4];
#pragma unroll
        for (int kc = 0; kc < 4; ++kc) {
            BFrag kb0, kb1;
#pragma unroll
            for (int jf = 0; jf < 8; ++jf) {
                kb0.u[jf] = *(const unsigned*)&Ks[kc * 16 + r16][kOffA(jf, half)];
                kb1.u[jf] =
                    *(const unsigned*)&Ks[kc * 16 + r16][32 + kOffA(jf, half)];
            }
            v8f a = vzero();
            a = wmma_bf16(qf[0].v, kb0.v, a);
            a = wmma_bf16(qf[1].v, kb1.v, a);
            s[kc] = a;
        }

        // ---- causal mask (wave-uniform branch; keeps EXEC full for WMMA) ----
        if (kB + 63 > qBase + w * 16) {
#pragma unroll
            for (int kc = 0; kc < 4; ++kc)
#pragma unroll
                for (int v = 0; v < 8; ++v) {
                    const int kpos = kB + kc * 16 + r16;
                    const int qp   = qBase + w * 16 + v + 8 * half;
                    if (kpos > qp) s[kc][v] = -1e30f;
                }
        }

        // ---- online softmax (row = v + 8*half; 16-lane shfl reductions) ----
#pragma unroll
        for (int v = 0; v < 8; ++v) {
            float r = fmaxf(fmaxf(s[0][v], s[1][v]), fmaxf(s[2][v], s[3][v]));
            r = fmaxf(r, __shfl_xor(r, 1, 16));
            r = fmaxf(r, __shfl_xor(r, 2, 16));
            r = fmaxf(r, __shfl_xor(r, 4, 16));
            r = fmaxf(r, __shfl_xor(r, 8, 16));
            const float mn = fmaxf(mrow[v], r);
            const float al = __expf(mrow[v] - mn);
            mrow[v] = mn;
            float rs = 0.f;
#pragma unroll
            for (int kc = 0; kc < 4; ++kc) {
                s[kc][v] = __expf(s[kc][v] - mn);
                rs += s[kc][v];
            }
            rs += __shfl_xor(rs, 1, 16);
            rs += __shfl_xor(rs, 2, 16);
            rs += __shfl_xor(rs, 4, 16);
            rs += __shfl_xor(rs, 8, 16);
            lrow[v] = lrow[v] * al + rs;
#pragma unroll
            for (int t = 0; t < 4; ++t) o[t][v] *= al;
        }

        // ---- C-layout -> A-layout via per-wave LDS staging of P ----
#pragma unroll
        for (int kc = 0; kc < 4; ++kc)
#pragma unroll
            for (int v = 0; v < 8; ++v)
                Ps[w][v + 8 * half][kc * 16 + r16] = (__bf16)s[kc][v];

        AFrag pf[2];
#pragma unroll
        for (int f = 0; f < 2; ++f)
#pragma unroll
            for (int jf = 0; jf < 8; ++jf) {
                const int key = 32 * f + kOffA(jf, half);
                pf[f].u[jf] = *(const unsigned*)&Ps[w][r16][key];
            }

        // ---- O += P V  (K-dim = 64 keys = 2x32; N = dhead in 4 tiles) ----
#pragma unroll
        for (int t = 0; t < 4; ++t) {
            BFrag vb0, vb1;
#pragma unroll
            for (int jf = 0; jf < 8; ++jf) {
                vb0.u[jf] = *(const unsigned*)&VT[t * 16 + r16][kOffA(jf, half)];
                vb1.u[jf] =
                    *(const unsigned*)&VT[t * 16 + r16][32 + kOffA(jf, half)];
            }
            o[t] = wmma_bf16(pf[0].v, vb0.v, o[t]);
            o[t] = wmma_bf16(pf[1].v, vb1.v, o[t]);
        }
    }

    // ---- normalize and scatter into y[b][pos][h*64+dh] (fp32) ----
#pragma unroll
    for (int t = 0; t < 4; ++t)
#pragma unroll
        for (int v = 0; v < 8; ++v) {
            const int pos = qBase + w * 16 + v + 8 * half;
            const int dh  = t * 16 + r16;
            const float val = o[t][v] / lrow[v];
            yws[((size_t)b * SEQ + pos) * D_MODEL + h * DHEAD + dh] = val;
        }
}

// ---------------------------------------------------------------------------
extern "C" void kernel_launch(void* const* d_in, const int* in_sizes, int n_in,
                              void* d_out, int out_size, void* d_ws,
                              size_t ws_size, hipStream_t stream) {
    const float* x    = (const float*)d_in[0];  // [4,2048,1024]
    const float* wqkv = (const float*)d_in[1];  // [3072,1024]
    const float* wout = (const float*)d_in[2];  // [1024,1024]
    float* out = (float*)d_out;                 // [4,2048,1024]

    char* ws = (char*)d_ws;
    __bf16* qws = (__bf16*)ws;                              // 16 MiB
    __bf16* kws = qws + QKV_ELEMS;                          // 16 MiB
    __bf16* vws = kws + QKV_ELEMS;                          // 16 MiB (transposed tiles)
    float*  yws = (float*)(ws + 3 * QKV_ELEMS * sizeof(__bf16));  // 32 MiB

    // 1) QKV projection + fused RoPE + scale; bf16 stores to [b,h,s,dh]
    gemm_bf16_wmma<1><<<dim3(3 * D_MODEL / 128, MTOT / 64), 256, 0, stream>>>(
        x, wqkv, nullptr, qws, kws, vws, MTOT, 3 * D_MODEL, D_MODEL);

    // 2) causal flash attention (TDM K staging + async V staging)
    flash_attn_wmma<<<dim3(SEQ / 64, NHEADS, BATCH), 128, 0, stream>>>(
        qws, kws, vws, yws);

    // 3) output projection
    gemm_bf16_wmma<0><<<dim3(D_MODEL / 128, MTOT / 64), 256, 0, stream>>>(
        yws, wout, out, nullptr, nullptr, nullptr, MTOT, D_MODEL, D_MODEL);
}
